// SGRU_69681549410660
// MI455X (gfx1250) — matrix operand order repaired and checked by reading, not compiled, not run on hardware
//
#include <hip/hip_runtime.h>

// ---------------------------------------------------------------------------
// Stacked GRU (4 stacks, bug-faithful: stack 3 reuses stack 2's weights),
// T=256 sequential steps. bf16 WMMA (v_wmma_f32_16x16x32_bf16) with f32
// accumulation. All GEMM operands are bf16 in global memory (weights
// converted once per launch; activations keep a bf16 mirror), so LDS tiles
// are staged with GLOBAL_LOAD_ASYNC_TO_LDS_B128 (ASYNCcnt) and the K loop is
// double-buffered: one s_wait_asynccnt + one barrier per K-step. K loops are
// phase-split so WMMA accumulator targets are compile-time fixed (no phi
// copies), and staging addresses are marched pointers (no per-step mul_u64).
// ---------------------------------------------------------------------------

#define IN_SZ 512
#define HD    1024
#define NST   4
#define BATCH 256
#define TLEN  256

#define BM 64          // block tile M (batch rows)
#define BN 64          // block tile N (output cols)
#define BK 32          // K step (matches WMMA K)
#define LDT 40         // padded LDS row stride (bf16): 80B, 16B-aligned,
                       // 20-bank stride -> conflict-free for 16-lane frags

typedef __attribute__((ext_vector_type(16))) __bf16 v16bf;
typedef __attribute__((ext_vector_type(8)))  float  v8f;
typedef __attribute__((ext_vector_type(4)))  int    v4i_t;
typedef __attribute__((address_space(1))) v4i_t g_v4i;   // global int4
typedef __attribute__((address_space(3))) v4i_t l_v4i;   // LDS int4

// ---------------- async copy global(bf16) -> LDS, 16B per lane --------------
__device__ __forceinline__ void async_b128(const void* g, void* l) {
#if __has_builtin(__builtin_amdgcn_global_load_async_to_lds_b128)
    __builtin_amdgcn_global_load_async_to_lds_b128(
        (g_v4i*)g, (l_v4i*)l, /*imm offset=*/0, /*cpol=*/0);
#else
    *(int4*)l = *(const int4*)g;   // synchronous fallback
#endif
}

__device__ __forceinline__ void wait_async0() {
#if __has_builtin(__builtin_amdgcn_s_wait_asynccnt)
    __builtin_amdgcn_s_wait_asynccnt(0);
#elif __has_builtin(__builtin_amdgcn_global_load_async_to_lds_b128)
    asm volatile("s_wait_asynccnt 0x0" ::: "memory");
#endif
}

// ---------------- WMMA fragment helpers (wave32, 16x16x32 bf16) ------------

// A fragment: 16x32 (MxK). Lane L: M = L&15; hi = L>>4 selects K interleave.
// Per ISA: VGPR j pairs -> K = 16*(j>>2) + 8*hi + 2*(j&3) + {0,1}
__device__ __forceinline__ v16bf load_frag_a(const __bf16* __restrict__ As,
                                             int row_base, int lane) {
    const int lo = lane & 15;
    const int hi = lane >> 4;
    const __bf16* r = As + (row_base + lo) * LDT;
    v16bf a;
#pragma unroll
    for (int j = 0; j < 8; ++j) {
        const int k = 16 * (j >> 2) + 8 * hi + 2 * (j & 3);
        a[2 * j]     = r[k];
        a[2 * j + 1] = r[k + 1];
    }
    return a;
}

// B fragment: 32x16 (KxN), stored in LDS as Bs[n][k] (column-major tile).
// Same K interleave as A: element pair j -> K = 16*(j>>2) + 8*hi + 2*(j&3); N = lane&15.
__device__ __forceinline__ v16bf load_frag_b(const __bf16* __restrict__ Bs,
                                             int col_base, int lane) {
    const int lo = lane & 15;
    const int hi = lane >> 4;
    const __bf16* c = Bs + (col_base + lo) * LDT;
    v16bf b;
#pragma unroll
    for (int j = 0; j < 8; ++j) {
        const int k = 16 * (j >> 2) + 8 * hi + 2 * (j & 3);
        b[2 * j]     = c[k];
        b[2 * j + 1] = c[k + 1];
    }
    return b;
}

__device__ __forceinline__ v8f wmma_bf16(v16bf a, v16bf b, v8f c) {
    return __builtin_amdgcn_wmma_f32_16x16x32_bf16(
        false, a, false, b, (short)0, c, false, false);
}

__device__ __forceinline__ float sigmoidf_(float x) {
    return 1.0f / (1.0f + __expf(-x));
}

// ---------------- fused GRU gate kernel ------------------------------------
// Phase 1 (16 K-steps): x @ Wx^T feeds u, r, xn.
// Phase 2 (32 K-steps): h @ Wh^T feeds u, r, hn.
// u/r accumulate BOTH GEMMs in one accumulator (they are summed anyway);
// n keeps xn/hn separate because r multiplies only hn.
__global__ __launch_bounds__(256) void gru_gate_kernel(
    const __bf16* __restrict__ x_bf,        // [B][IN_SZ]
    const float*  __restrict__ h_in_f,      // [B][NST*HD] (+ stack offset)
    const __bf16* __restrict__ h_in_bf,     // [B][NST*HD] (+ stack offset)
    float*        __restrict__ h_out_f,
    __bf16*       __restrict__ h_out_bf,
    const __bf16* __restrict__ Wx,          // [3*HD][IN_SZ]
    const __bf16* __restrict__ Wh,          // [3*HD][HD]
    const float*  __restrict__ bx,          // [3*HD]
    const float*  __restrict__ bh)          // [3*HD]
{
    __shared__ __bf16 As[2][BM * LDT];
    __shared__ __bf16 Bs[2][3][BN * LDT];

    const int tid  = threadIdx.x;
    const int lane = tid & 31;
    const int wave = tid >> 5;
    const int wm   = wave & 3;     // 4 x 16 rows
    const int wn   = wave >> 2;    // 2 x 32 cols
    const int bm0  = blockIdx.x * BM;
    const int n0   = blockIdx.y * BN;

    const int srow = tid >> 2;        // 0..63 staging row
    const int sk   = (tid & 3) * 8;   // 0,8,16,24 (bf16 elems; 16B chunks)
    const int HSTR = NST * HD;

    v8f accU[2], accR[2], accXN[2], accHN[2];
#pragma unroll
    for (int tn = 0; tn < 2; ++tn) {
        accU[tn] = {}; accR[tn] = {}; accXN[tn] = {}; accHN[tn] = {};
    }

    // per-thread marching staging pointers (advance by BK per K-step)
    const __bf16* aP = x_bf + (size_t)(bm0 + srow) * IN_SZ + sk;
    const __bf16* w0 = Wx + (size_t)(0 * HD + n0 + srow) * IN_SZ + sk;
    const __bf16* w1 = Wx + (size_t)(1 * HD + n0 + srow) * IN_SZ + sk;
    const __bf16* w2 = Wx + (size_t)(2 * HD + n0 + srow) * IN_SZ + sk;

    __bf16* const aD  = &As[0][srow * LDT + sk];
    __bf16* const bD0 = &Bs[0][0][srow * LDT + sk];
    __bf16* const bD1 = &Bs[0][1][srow * LDT + sk];
    __bf16* const bD2 = &Bs[0][2][srow * LDT + sk];
    const int bufOff = BM * LDT;               // As buffer stride (elems)
    const int bufOffB = 3 * BN * LDT;          // Bs buffer stride (elems)

    auto issue = [&](int buf) {
        async_b128(aP, aD  + buf * bufOff);
        async_b128(w0, bD0 + buf * bufOffB);
        async_b128(w1, bD1 + buf * bufOffB);
        async_b128(w2, bD2 + buf * bufOffB);
    };

    issue(0);
    // -------- Phase 1: 16 K-steps over x ----------------------------------
    for (int kk = 0; kk < 16; ++kk) {
        wait_async0();          // this wave's copies for step kk have landed
        __syncthreads();        // all waves' copies landed; prev reads done
        if (kk < 15) {
            aP += BK; w0 += BK; w1 += BK; w2 += BK;
        } else {                // re-point at phase-2 sources (step 16)
            aP = h_in_bf + (size_t)(bm0 + srow) * HSTR + sk;
            w0 = Wh + (size_t)(0 * HD + n0 + srow) * HD + sk;
            w1 = Wh + (size_t)(1 * HD + n0 + srow) * HD + sk;
            w2 = Wh + (size_t)(2 * HD + n0 + srow) * HD + sk;
        }
        issue((kk + 1) & 1);

        const int cur = kk & 1;
        const v16bf a = load_frag_a(As[cur], wm * 16, lane);
#pragma unroll
        for (int tn = 0; tn < 2; ++tn) {
            const int cb = wn * 32 + tn * 16;
            const v16bf bu = load_frag_b(Bs[cur][0], cb, lane);
            const v16bf br = load_frag_b(Bs[cur][1], cb, lane);
            const v16bf bn = load_frag_b(Bs[cur][2], cb, lane);
            accU[tn]  = wmma_bf16(a, bu, accU[tn]);
            accR[tn]  = wmma_bf16(a, br, accR[tn]);
            accXN[tn] = wmma_bf16(a, bn, accXN[tn]);
        }
    }
    // -------- Phase 2: 32 K-steps over h ----------------------------------
    for (int kk = 16; kk < 48; ++kk) {
        wait_async0();
        __syncthreads();
        if (kk < 47) {
            aP += BK; w0 += BK; w1 += BK; w2 += BK;
            issue((kk + 1) & 1);
        }
        const int cur = kk & 1;
        const v16bf a = load_frag_a(As[cur], wm * 16, lane);
#pragma unroll
        for (int tn = 0; tn < 2; ++tn) {
            const int cb = wn * 32 + tn * 16;
            const v16bf bu = load_frag_b(Bs[cur][0], cb, lane);
            const v16bf br = load_frag_b(Bs[cur][1], cb, lane);
            const v16bf bn = load_frag_b(Bs[cur][2], cb, lane);
            accU[tn]  = wmma_bf16(a, bu, accU[tn]);
            accR[tn]  = wmma_bf16(a, br, accR[tn]);
            accHN[tn] = wmma_bf16(a, bn, accHN[tn]);
        }
    }

    // Epilogue. C layout: VGPR r: lanes 0-15 -> M = r; 16-31 -> M = 8 + r.
    const int lo = lane & 15;
    const int hi = lane >> 4;
#pragma unroll
    for (int tn = 0; tn < 2; ++tn) {
        const int c   = n0 + wn * 32 + tn * 16 + lo;
        const float bu  = bx[c] + bh[c];
        const float br  = bx[HD + c] + bh[HD + c];
        const float bxn = bx[2 * HD + c];
        const float bhn = bh[2 * HD + c];
#pragma unroll
        for (int r = 0; r < 8; ++r) {
            const int row = bm0 + wm * 16 + hi * 8 + r;
            const float u  = sigmoidf_(accU[tn][r] + bu);
            const float rr = sigmoidf_(accR[tn][r] + br);
            const float nn = tanhf(accXN[tn][r] + bxn + rr * (accHN[tn][r] + bhn));
            const float ho = h_in_f[(size_t)row * HSTR + c];
            const float hn = u * ho + (1.0f - u) * nn;
            h_out_f[(size_t)row * HSTR + c]  = hn;
            h_out_bf[(size_t)row * HSTR + c] = (__bf16)hn;
        }
    }
}

// ---------------- output projection: cat(x, h_new) @ Wo^T + bo -------------
__global__ __launch_bounds__(256) void gru_out_kernel(
    const __bf16* __restrict__ x_bf,        // [B][IN_SZ]
    const __bf16* __restrict__ h_bf,        // [B][NST*HD] (+ stack offset)
    const __bf16* __restrict__ Wo,          // [IN_SZ][IN_SZ+HD]
    const float*  __restrict__ bo,          // [IN_SZ]
    float*        __restrict__ out_f,       // [B][IN_SZ]
    __bf16*       __restrict__ out_bf)      // [B][IN_SZ]
{
    __shared__ __bf16 As[2][BM * LDT];
    __shared__ __bf16 Bs[2][BN * LDT];

    const int tid  = threadIdx.x;
    const int lane = tid & 31;
    const int wave = tid >> 5;
    const int wm   = wave & 3;
    const int wn   = wave >> 2;
    const int bm0  = blockIdx.x * BM;
    const int n0   = blockIdx.y * BN;

    const int srow = tid >> 2;
    const int sk   = (tid & 3) * 8;
    const int HSTR = NST * HD;
    const int KTOT = IN_SZ + HD;

    v8f acc[2];
    acc[0] = {}; acc[1] = {};

    const __bf16* aP = x_bf + (size_t)(bm0 + srow) * IN_SZ + sk;
    const __bf16* wP = Wo + (size_t)(n0 + srow) * KTOT + sk;

    __bf16* const aD = &As[0][srow * LDT + sk];
    __bf16* const bD = &Bs[0][srow * LDT + sk];
    const int bufOffA = BM * LDT;
    const int bufOffB = BN * LDT;

    auto issue = [&](int buf) {
        async_b128(aP, aD + buf * bufOffA);
        async_b128(wP, bD + buf * bufOffB);
    };

    issue(0);
    // -------- Phase 1: 16 K-steps over x (Wo cols 0..511) ------------------
    for (int kk = 0; kk < 16; ++kk) {
        wait_async0();
        __syncthreads();
        wP += BK;
        if (kk < 15) aP += BK;
        else         aP = h_bf + (size_t)(bm0 + srow) * HSTR + sk;
        issue((kk + 1) & 1);

        const int cur = kk & 1;
        const v16bf a = load_frag_a(As[cur], wm * 16, lane);
#pragma unroll
        for (int tn = 0; tn < 2; ++tn) {
            const v16bf b = load_frag_b(Bs[cur], wn * 32 + tn * 16, lane);
            acc[tn] = wmma_bf16(a, b, acc[tn]);
        }
    }
    // -------- Phase 2: 32 K-steps over h (Wo cols 512..1535) ---------------
    for (int kk = 16; kk < 48; ++kk) {
        wait_async0();
        __syncthreads();
        if (kk < 47) {
            aP += BK; wP += BK;
            issue((kk + 1) & 1);
        }
        const int cur = kk & 1;
        const v16bf a = load_frag_a(As[cur], wm * 16, lane);
#pragma unroll
        for (int tn = 0; tn < 2; ++tn) {
            const v16bf b = load_frag_b(Bs[cur], wn * 32 + tn * 16, lane);
            acc[tn] = wmma_bf16(a, b, acc[tn]);
        }
    }

    const int lo = lane & 15;
    const int hi = lane >> 4;
#pragma unroll
    for (int tn = 0; tn < 2; ++tn) {
        const int c  = n0 + wn * 32 + tn * 16 + lo;
        const float bb = bo[c];
#pragma unroll
        for (int r = 0; r < 8; ++r) {
            const int row = bm0 + wm * 16 + hi * 8 + r;
            const float v = acc[tn][r] + bb;
            out_f[(size_t)row * IN_SZ + c]  = v;
            out_bf[(size_t)row * IN_SZ + c] = (__bf16)v;
        }
    }
}

// ---------------- log_softmax over 512 cols, one block per row -------------
__global__ __launch_bounds__(256) void logsoftmax_kernel(
    const float* __restrict__ in, float* __restrict__ out, int out_stride)
{
    __shared__ float red[256];
    const int b   = blockIdx.x;
    const int tid = threadIdx.x;
    const float v0 = in[(size_t)b * IN_SZ + tid];
    const float v1 = in[(size_t)b * IN_SZ + 256 + tid];

    red[tid] = fmaxf(v0, v1);
    __syncthreads();
    for (int s = 128; s > 0; s >>= 1) {
        if (tid < s) red[tid] = fmaxf(red[tid], red[tid + s]);
        __syncthreads();
    }
    const float m = red[0];
    __syncthreads();

    red[tid] = __expf(v0 - m) + __expf(v1 - m);
    __syncthreads();
    for (int s = 128; s > 0; s >>= 1) {
        if (tid < s) red[tid] += red[tid + s];
        __syncthreads();
    }
    const float lse = m + __logf(red[0]);

    out[(size_t)b * out_stride + tid]       = v0 - lse;
    out[(size_t)b * out_stride + 256 + tid] = v1 - lse;
}

// ---------------- utility kernels ------------------------------------------
__global__ void f32_to_bf16_kernel(const float* __restrict__ in,
                                   __bf16* __restrict__ out, int n) {
    int i = blockIdx.x * blockDim.x + threadIdx.x;
    const int stride = gridDim.x * blockDim.x;
    for (; i < n; i += stride) out[i] = (__bf16)in[i];
}

__global__ void copy_f32_kernel(const float* __restrict__ in,
                                float* __restrict__ out, int n) {
    int i = blockIdx.x * blockDim.x + threadIdx.x;
    const int stride = gridDim.x * blockDim.x;
    for (; i < n; i += stride) out[i] = in[i];
}

// convert the strided time-slice x[:, t, :] (f32) into a dense bf16 [B][IN_SZ]
__global__ __launch_bounds__(256) void xslice_bf16_kernel(
    const float* __restrict__ src, __bf16* __restrict__ dst) {
    const int i = blockIdx.x * blockDim.x + threadIdx.x;  // 0 .. B*IN_SZ-1
    const int b = i >> 9, c = i & (IN_SZ - 1);
    dst[i] = (__bf16)src[(size_t)b * (TLEN * IN_SZ) + c];
}

// ---------------------------------------------------------------------------
extern "C" void kernel_launch(void* const* d_in, const int* in_sizes, int n_in,
                              void* d_out, int out_size, void* d_ws, size_t ws_size,
                              hipStream_t stream) {
    (void)in_sizes; (void)n_in; (void)out_size; (void)ws_size;

    const float* xb = (const float*)d_in[0];  // [B][T][IN_SZ]
    const float* h0 = (const float*)d_in[1];  // [B][NST][HD]
    const float* Wx = (const float*)d_in[2];  // [NST][3*HD][IN_SZ]
    const float* bx = (const float*)d_in[3];  // [NST][3*HD]
    const float* Wh = (const float*)d_in[4];  // [NST][3*HD][HD]
    const float* bh = (const float*)d_in[5];  // [NST][3*HD]
    const float* Wo = (const float*)d_in[6];  // [NST][IN_SZ][IN_SZ+HD]
    const float* bo = (const float*)d_in[7];  // [NST][IN_SZ]

    float* preds_out = (float*)d_out;                            // [B][T][IN_SZ]
    float* h_final   = preds_out + (size_t)BATCH * TLEN * IN_SZ; // [B][NST][HD]

    // ---- workspace carve-up (bf16 blocks first, all 16B aligned) ----
    const size_t nWx = (size_t)NST * 3 * HD * IN_SZ;
    const size_t nWh = (size_t)NST * 3 * HD * HD;
    const size_t nWo = (size_t)NST * IN_SZ * (IN_SZ + HD);
    const size_t nH  = (size_t)BATCH * NST * HD;
    const size_t nP  = (size_t)BATCH * IN_SZ;

    __bf16* Wxc  = (__bf16*)d_ws;
    __bf16* Whc  = Wxc + nWx;
    __bf16* Woc  = Whc + nWh;
    __bf16* hb0  = Woc + nWo;       // bf16 hidden mirrors (ping-pong)
    __bf16* hb1  = hb0 + nH;
    __bf16* pb0  = hb1 + nH;        // bf16 pred mirrors (ping-pong)
    __bf16* pb1  = pb0 + nP;
    __bf16* xbf  = pb1 + nP;        // bf16 slice of xb[:, t, :]
    float*  hf0  = (float*)(xbf + nP);
    float*  hf1  = hf0 + nH;
    float*  pf0  = hf1 + nH;
    float*  pf1  = pf0 + nP;

    __bf16* hbs[2] = {hb0, hb1};
    float*  hfs[2] = {hf0, hf1};

    // ---- one-time (per launch) conversions ----
    f32_to_bf16_kernel<<<2048, 256, 0, stream>>>(Wx, Wxc, (int)nWx);
    f32_to_bf16_kernel<<<2048, 256, 0, stream>>>(Wh, Whc, (int)nWh);
    f32_to_bf16_kernel<<<2048, 256, 0, stream>>>(Wo, Woc, (int)nWo);
    f32_to_bf16_kernel<<<2048, 256, 0, stream>>>(h0, hb0, (int)nH);
    copy_f32_kernel<<<2048, 256, 0, stream>>>(h0, hf0, (int)nH);

    const dim3 blk(256);
    const dim3 gate_grid(BATCH / BM, HD / BN);      // 4 x 16
    const dim3 out_grid(BATCH / BM, IN_SZ / BN);    // 4 x 8

    for (int t = 0; t < TLEN; ++t) {
        float*  cur_hf = hfs[t & 1];
        float*  nxt_hf = hfs[(t + 1) & 1];
        __bf16* cur_hb = hbs[t & 1];
        __bf16* nxt_hb = hbs[(t + 1) & 1];

        xslice_bf16_kernel<<<(BATCH * IN_SZ) / 256, blk, 0, stream>>>(
            xb + (size_t)t * IN_SZ, xbf);

        const __bf16* xin = xbf;
        __bf16* pbs[2] = {pb0, pb1};
        float*  pfs[2] = {pf0, pf1};
        int pi = 0;
        const float* last_pred = nullptr;

        for (int s = 0; s < NST; ++s) {
            const int w = (s < NST - 1) ? s : (NST - 2);  // bug-faithful reuse

            gru_gate_kernel<<<gate_grid, blk, 0, stream>>>(
                xin,
                cur_hf + (size_t)s * HD, cur_hb + (size_t)s * HD,
                nxt_hf + (size_t)s * HD, nxt_hb + (size_t)s * HD,
                Wxc + (size_t)w * 3 * HD * IN_SZ,
                Whc + (size_t)w * 3 * HD * HD,
                bx + (size_t)w * 3 * HD,
                bh + (size_t)w * 3 * HD);

            gru_out_kernel<<<out_grid, blk, 0, stream>>>(
                xin,
                nxt_hb + (size_t)s * HD,
                Woc + (size_t)w * IN_SZ * (IN_SZ + HD),
                bo + (size_t)w * IN_SZ,
                pfs[pi], pbs[pi]);

            xin = pbs[pi];
            last_pred = pfs[pi];
            pi ^= 1;
        }

        logsoftmax_kernel<<<BATCH, blk, 0, stream>>>(
            last_pred, preds_out + (size_t)t * IN_SZ, TLEN * IN_SZ);
    }

    // final hidden state: t = TLEN-1 wrote hfs[TLEN & 1]; copy it out.
    copy_f32_kernel<<<2048, 256, 0, stream>>>(hfs[TLEN & 1], h_final, (int)nH);
}
